// LSTMCell_25142738550992
// MI455X (gfx1250) — matrix operand order repaired and checked by reading, not compiled, not run on hardware
//
#include <hip/hip_runtime.h>

// ---------------------------------------------------------------------------
// LSTM cell, fused: gates = [x|h] @ [Wx|Wh]^T + bx + bh ; gated elementwise.
// B=32768, I=256, H=512, K=I+H=768, 4H=2048.
// Strategy: bf16 WMMA (v_wmma_f32_16x16x32_bf16) with f32 accumulate,
// fragment-linear packed operands in workspace, ASYNC global->LDS staging
// (ASYNCcnt) with TRIPLE buffering so global latency overlaps WMMA work,
// spill-free LDS gate exchange + fused epilogue (rcp/exp gates) -> ct, ht.
// ---------------------------------------------------------------------------

typedef __attribute__((ext_vector_type(16))) __bf16 v16bf;
typedef __attribute__((ext_vector_type(8)))  float  v8f;

#define B_   32768
#define I_   256
#define H_   512
#define K_   768        // I_ + H_
#define KS_  24         // K_ / 32 (k-steps of 32)
#define GN_  2048       // 4 * H_
#define BH_  (B_ * H_)  // 16,777,216

// fragment-linear sizes
#define A_TILES   (B_ / 16)          // 2048 row-tiles of A
#define W_TILES   (GN_ / 16)         // 128 row-tiles of W
#define A_FRAGS   (A_TILES * KS_ * 32)   // lane-chunks (32B each)
#define W_FRAGS   (W_TILES * KS_ * 32)
#define ASW_BYTES ((size_t)A_FRAGS * 32) // 48 MB

#define STAGE_U4   1536                  // uint4 per stage buffer (24 KB)
#define STAGE_B    (STAGE_U4 * 16)
#define LDS_BYTES  (3 * STAGE_B)         // 72 KB (also covers 64KB exchange)

__device__ __forceinline__ unsigned short f2bf(float f) {
    unsigned u = __float_as_uint(f);
    unsigned r = u + 0x7FFFu + ((u >> 16) & 1u);   // round-to-nearest-even
    return (unsigned short)(r >> 16);
}

template <int N>
__device__ __forceinline__ void wait_async_c() {
#if __has_builtin(__builtin_amdgcn_s_wait_asynccnt)
    __builtin_amdgcn_s_wait_asynccnt(N);
#else
    asm volatile("s_wait_asynccnt %0" :: "i"(N) : "memory");
#endif
}

// Async copy of one 16-byte chunk: global -> LDS, tracked by ASYNCcnt.
__device__ __forceinline__ void async_copy16(unsigned lds_byte_addr, const void* gptr) {
    asm volatile("global_load_async_to_lds_b128 %0, %1, off"
                 :: "v"(lds_byte_addr), "v"(gptr)
                 : "memory");
}

__device__ __forceinline__ float fast_sigmoid(float x) {
    return __builtin_amdgcn_rcpf(1.0f + __expf(-x));
}
__device__ __forceinline__ float fast_tanh(float x) {
    // tanh(x) = 1 - 2/(exp(2x)+1); saturates correctly for |x| large.
    return 1.0f - 2.0f * __builtin_amdgcn_rcpf(__expf(2.0f * x) + 1.0f);
}

// ---------------------------------------------------------------------------
// Pack pre-pass: f32 -> bf16, fragment-linear layout.
//   A lane layout (16x32 MxK tile): lane l holds row (l&15),
//       K = kb..kb+7 and kb+16..kb+23, kb = ks*32 + (l>>4)*8.
//   B lane layout (32x16 KxN tile, W row-major [N,K]): lane l holds col (l&15),
//       K = kb..kb+7 and kb+16..kb+23, kb = ks*32 + (l>>4)*8.
// One thread produces one 32B lane-chunk (16 bf16).
// ---------------------------------------------------------------------------
__global__ __launch_bounds__(256) void pack_bf16_kernel(
    const float* __restrict__ x,  const float* __restrict__ h,
    const float* __restrict__ Wx, const float* __restrict__ Wh,
    unsigned short* __restrict__ Asw, unsigned short* __restrict__ Wsw)
{
    int idx = blockIdx.x * 256 + threadIdx.x;
    if (idx >= A_FRAGS + W_FRAGS) return;

    bool isA = idx < A_FRAGS;
    int li   = isA ? idx : (idx - A_FRAGS);
    int lane = li & 31;
    int ks   = (li >> 5) % KS_;
    int t    = li / (32 * KS_);

    int row = t * 16 + (lane & 15);
    int hf  = lane >> 4;
    int c0, c1;
    if (isA) { int kb = ks * 32 + hf * 8;  c0 = kb; c1 = kb + 16; }
    else     { int kb = ks * 32 + hf * 8;  c0 = kb; c1 = kb + 16; }

    float vals[16];
    int starts[2] = { c0, c1 };
#pragma unroll
    for (int c = 0; c < 2; c++) {
        int kk = starts[c];
        const float* src;
        if (isA) src = (kk < I_) ? (x  + (size_t)row * I_ + kk)
                                 : (h  + (size_t)row * H_ + (kk - I_));
        else     src = (kk < I_) ? (Wx + (size_t)row * I_ + kk)
                                 : (Wh + (size_t)row * H_ + (kk - I_));
        float4 p0 = *(const float4*)(src);
        float4 p1 = *(const float4*)(src + 4);
        vals[c * 8 + 0] = p0.x; vals[c * 8 + 1] = p0.y;
        vals[c * 8 + 2] = p0.z; vals[c * 8 + 3] = p0.w;
        vals[c * 8 + 4] = p1.x; vals[c * 8 + 5] = p1.y;
        vals[c * 8 + 6] = p1.z; vals[c * 8 + 7] = p1.w;
    }

    unsigned int w[8];
#pragma unroll
    for (int i = 0; i < 8; i++)
        w[i] = (unsigned)f2bf(vals[2 * i]) | ((unsigned)f2bf(vals[2 * i + 1]) << 16);

    unsigned short* dstbase =
        (isA ? Asw : Wsw) + ((size_t)(t * KS_ + ks) * 32 + lane) * 16;
    uint4* dp = (uint4*)dstbase;
    dp[0] = make_uint4(w[0], w[1], w[2], w[3]);
    dp[1] = make_uint4(w[4], w[5], w[6], w[7]);
}

// ---------------------------------------------------------------------------
// Fused GEMM + LSTM gating.
// Grid: (B/128, H/64). Block: 256 threads = 8 wave32.
// Wave w: pm = w&3 (m-tile pair), gp = w>>2 (gate pair {0,1} or {2,3}).
// Accumulators: acc[mi][gi][nt] (2 m x 2 gates x 4 n), 16x16 f32 tiles.
// Dynamic LDS: 3 x 24KB stage buffers; epilogue reuses 64KB for exchange.
// ---------------------------------------------------------------------------
__global__ __launch_bounds__(256) void lstm_cell_wmma_kernel(
    const uint4* __restrict__ Asw, const uint4* __restrict__ Wsw,
    const float* __restrict__ bx,  const float* __restrict__ bh,
    const float* __restrict__ cin, float* __restrict__ outp)
{
    extern __shared__ uint4 smem[];   // LDS_BYTES = 72 KB

    const int tid  = threadIdx.x;
    const int lane = tid & 31;
    const int wave = tid >> 5;
    const int pm   = wave & 3;
    const int gp   = wave >> 2;
    const int m0   = blockIdx.x * 128;
    const int j0   = blockIdx.y * 64;

    // ---- staging plan: 24 tiles x 1KB per k-step = 1536 x 16B; 6 per thread
    const uint4* src[6];
    unsigned     ldst[6];      // LDS byte address of chunk in buffer 0
    const unsigned smemBase = (unsigned)(size_t)(const void*)&smem[0];
#pragma unroll
    for (int i = 0; i < 6; i++) {
        int sid  = tid + i * 256;
        int tile = sid >> 6;
        int off  = sid & 63;
        const uint4* base;
        if (tile < 8) {
            base = Asw + (size_t)((m0 / 16 + tile) * KS_) * 64;
        } else {
            int b  = tile - 8;
            int g  = b >> 2;
            int nt = b & 3;
            base = Wsw + (size_t)((g * (H_ / 16) + j0 / 16 + nt) * KS_) * 64;
        }
        src[i]  = base + off;
        ldst[i] = smemBase + (unsigned)((tile * 64 + off) * 16);
    }

    v8f acc[2][2][4];
    const v8f vzero = { 0.f, 0.f, 0.f, 0.f, 0.f, 0.f, 0.f, 0.f };
#pragma unroll
    for (int mi = 0; mi < 2; mi++)
#pragma unroll
        for (int gi = 0; gi < 2; gi++)
#pragma unroll
            for (int nt = 0; nt < 4; nt++)
                acc[mi][gi][nt] = vzero;

    // One k-step of WMMA work from stage buffer starting at uint4 index `cur`.
    auto compute_step = [&](int cur) __attribute__((always_inline)) {
        v16bf a0 = *(const v16bf*)&smem[cur + (pm * 2 + 0) * 64 + lane * 2];
        v16bf a1 = *(const v16bf*)&smem[cur + (pm * 2 + 1) * 64 + lane * 2];
#pragma unroll
        for (int gi = 0; gi < 2; gi++) {
            const int g = gp * 2 + gi;
            v16bf b0 = *(const v16bf*)&smem[cur + (8 + g * 4 + 0) * 64 + lane * 2];
            v16bf b1 = *(const v16bf*)&smem[cur + (8 + g * 4 + 1) * 64 + lane * 2];
            v16bf b2 = *(const v16bf*)&smem[cur + (8 + g * 4 + 2) * 64 + lane * 2];
            v16bf b3 = *(const v16bf*)&smem[cur + (8 + g * 4 + 3) * 64 + lane * 2];

            acc[0][gi][0] = __builtin_amdgcn_wmma_f32_16x16x32_bf16(
                false, a0, false, b0, (short)0, acc[0][gi][0], false, false);
            acc[0][gi][1] = __builtin_amdgcn_wmma_f32_16x16x32_bf16(
                false, a0, false, b1, (short)0, acc[0][gi][1], false, false);
            acc[0][gi][2] = __builtin_amdgcn_wmma_f32_16x16x32_bf16(
                false, a0, false, b2, (short)0, acc[0][gi][2], false, false);
            acc[0][gi][3] = __builtin_amdgcn_wmma_f32_16x16x32_bf16(
                false, a0, false, b3, (short)0, acc[0][gi][3], false, false);
            acc[1][gi][0] = __builtin_amdgcn_wmma_f32_16x16x32_bf16(
                false, a1, false, b0, (short)0, acc[1][gi][0], false, false);
            acc[1][gi][1] = __builtin_amdgcn_wmma_f32_16x16x32_bf16(
                false, a1, false, b1, (short)0, acc[1][gi][1], false, false);
            acc[1][gi][2] = __builtin_amdgcn_wmma_f32_16x16x32_bf16(
                false, a1, false, b2, (short)0, acc[1][gi][2], false, false);
            acc[1][gi][3] = __builtin_amdgcn_wmma_f32_16x16x32_bf16(
                false, a1, false, b3, (short)0, acc[1][gi][3], false, false);
        }
    };

    // ---- prologue: async-stage k-steps 0 and 1 into buffers 0 and 1
#pragma unroll
    for (int i = 0; i < 6; i++)
        async_copy16(ldst[i], (const void*)src[i]);
#pragma unroll
    for (int i = 0; i < 6; i++)
        async_copy16(ldst[i] + STAGE_B, (const void*)(src[i] + 64));
    wait_async_c<6>();      // k-step 0 landed; k-step 1 may be in flight
    __syncthreads();

    // ---- steady state: ks = 0..KS_-3, triple-buffered
    int cb = 0;             // buffer holding k-step ks
    for (int ks = 0; ks < KS_ - 2; ks++) {
        int b2 = cb + 2; if (b2 >= 3) b2 -= 3;          // (ks+2) % 3
        const unsigned dstByte = (unsigned)(b2 * STAGE_B);
#pragma unroll
        for (int i = 0; i < 6; i++)
            async_copy16(ldst[i] + dstByte,
                         (const void*)(src[i] + (size_t)(ks + 2) * 64));

        compute_step(cb * STAGE_U4);

        wait_async_c<6>();  // ks+1 resident; ks+2 batch may stay in flight
        __syncthreads();
        cb = cb + 1; if (cb >= 3) cb -= 3;
    }

    // ---- tail: ks = KS_-2 (wait all), ks = KS_-1
    compute_step(cb * STAGE_U4);
    wait_async_c<0>();      // last batch resident
    __syncthreads();
    cb = cb + 1; if (cb >= 3) cb -= 3;

    compute_step(cb * STAGE_U4);
    __syncthreads();        // stage reads done; LDS reusable for exchange

    // ---- fused epilogue -----------------------------------------------------
    // Uniform LDS exchange: per round r (n-halves), every wave stores its 8
    // gate tiles (2m x 2g x 2n = 64KB total), then wave w combines all 4 gates
    // for m-tile w and writes ct/ht. No conditional register selection.
    const int mBase = m0 + wave * 16;    // output m-tile = wave id
    const int hf8   = (lane >> 4) * 8;
    const int jc    = lane & 15;

#pragma unroll
    for (int r = 0; r < 2; r++) {
#pragma unroll
        for (int mi = 0; mi < 2; mi++)
#pragma unroll
            for (int gi = 0; gi < 2; gi++)
#pragma unroll
                for (int ntl = 0; ntl < 2; ntl++) {
                    int slot = (((pm * 2 + mi) * 4 + (gp * 2 + gi)) * 2 + ntl); // 0..63
                    *(v8f*)&smem[slot * 64 + lane * 2] = acc[mi][gi][r * 2 + ntl];
                }
        __syncthreads();

#pragma unroll
        for (int ntl = 0; ntl < 2; ntl++) {
            int nt = r * 2 + ntl;
            int j  = j0 + nt * 16 + jc;

            v8f gI = *(const v8f*)&smem[((wave * 4 + 0) * 2 + ntl) * 64 + lane * 2];
            v8f gF = *(const v8f*)&smem[((wave * 4 + 1) * 2 + ntl) * 64 + lane * 2];
            v8f gO = *(const v8f*)&smem[((wave * 4 + 2) * 2 + ntl) * 64 + lane * 2];
            v8f gC = *(const v8f*)&smem[((wave * 4 + 3) * 2 + ntl) * 64 + lane * 2];

            float bs0 = bx[0 * H_ + j] + bh[0 * H_ + j];
            float bs1 = bx[1 * H_ + j] + bh[1 * H_ + j];
            float bs2 = bx[2 * H_ + j] + bh[2 * H_ + j];
            float bs3 = bx[3 * H_ + j] + bh[3 * H_ + j];

#pragma unroll
            for (int v = 0; v < 8; v++) {
                int m = mBase + v + hf8;
                int o = m * H_ + j;
                float it = fast_sigmoid(gI[v] + bs0);
                float ft = fast_sigmoid(gF[v] + bs1);
                float ot = fast_sigmoid(gO[v] + bs2);
                float cn = ft * cin[o] + it * fast_tanh(gC[v] + bs3);
                float hn = ot * fast_tanh(cn);
                outp[o]       = cn;
                outp[BH_ + o] = hn;
            }
        }
        __syncthreads();
    }
}

// ---------------------------------------------------------------------------
extern "C" void kernel_launch(void* const* d_in, const int* in_sizes, int n_in,
                              void* d_out, int out_size, void* d_ws, size_t ws_size,
                              hipStream_t stream)
{
    (void)in_sizes; (void)n_in; (void)out_size; (void)ws_size;

    const float* x  = (const float*)d_in[0];
    const float* h  = (const float*)d_in[1];
    const float* c  = (const float*)d_in[2];
    const float* Wx = (const float*)d_in[3];
    const float* bx = (const float*)d_in[4];
    const float* Wh = (const float*)d_in[5];
    const float* bh = (const float*)d_in[6];
    float* outp = (float*)d_out;

    unsigned short* Asw = (unsigned short*)d_ws;
    unsigned short* Wsw = (unsigned short*)((char*)d_ws + ASW_BYTES);

    int packThreads = A_FRAGS + W_FRAGS;            // 1,671,168
    int packBlocks  = (packThreads + 255) / 256;    // 6528

    hipLaunchKernelGGL(pack_bf16_kernel, dim3(packBlocks), dim3(256), 0, stream,
                       x, h, Wx, Wh, Asw, Wsw);

    hipLaunchKernelGGL(lstm_cell_wmma_kernel, dim3(B_ / 128, H_ / 64), dim3(256),
                       LDS_BYTES, stream,
                       (const uint4*)Asw, (const uint4*)Wsw, bx, bh, c, outp);
}
